// BandSplitModule_22591527977185
// MI455X (gfx1250) — matrix operand order repaired and checked by reading, not compiled, not run on hardware
//
#include <hip/hip_runtime.h>

// BandSplit fused kernel for gfx1250 (MI455X), round 2.
// Bandwidth-bound (~385 MB, ~8.4 GFLOP): read x once (vectorized float2),
// write out once, keep per-band W hot in L2, per-band D->128 GEMM via
// v_wmma_f32_16x16x32_f16 with f32 accumulate. All staging is branchless and
// vectorized (b128 global loads, b64 ds stores) so no serialized load loops.

typedef _Float16 v16h __attribute__((ext_vector_type(16)));
typedef float    v8f  __attribute__((ext_vector_type(8)));

#define BB     8
#define CC     2
#define FFREQ  1025
#define TT     1000
#define RR     2
#define EMBD   128
#define NBANDS 61
#define MROWS  (BB * TT)     // 8000 rows (b,t)
#define RPB    32            // rows per block = two 16-row WMMA M-tiles
#define EPSV   1e-5f

struct BSParams {
    const float* x;
    const float* gamma[5];
    const float* beta[5];
    const float* W[5];
    const float* bias[5];
    float* out;
};

union FragH { v16h v; unsigned u[8]; _Float16 h[16]; };
union Pack4 { _Float16 h[4]; uint2 u2; };

__global__ __launch_bounds__(256)
void bandsplit_wmma_kernel(BSParams p) {
    __shared__ _Float16 s_A[RPB * 256];   // 16 KB: raw f16, normalized in place
    __shared__ _Float16 s_B[128 * 32];    //  8 KB: W^T chunk for current K-step

    const int tid  = threadIdx.x;
    const int lane = tid & 31;            // wave32 lane
    const int wave = tid >> 5;            // 8 waves: wave owns N in [16w,16w+16)
    const int rloc = tid >> 3;            // local row 0..31 (8 threads per row)
    const int l8   = tid & 7;

    const int band  = blockIdx.y;
    const int mtile = blockIdx.x;

    // band -> (group, band-in-group, width, freq start)
    int g, bg, w, fstart;
    if (band < 32)      { g = 0; bg = band;      w = 8;  fstart = 0;    }
    else if (band < 48) { g = 1; bg = band - 32; w = 16; fstart = 256;  }
    else if (band < 56) { g = 2; bg = band - 48; w = 32; fstart = 512;  }
    else if (band < 60) { g = 3; bg = band - 56; w = 64; fstart = 768;  }
    else                { g = 4; bg = 0;         w = 1;  fstart = 1024; }
    const int D      = 4 * w;                  // LayerNorm/Linear input dim
    const int Kp     = (D < 32) ? 32 : D;      // K padded to WMMA step
    const int nsteps = Kp >> 5;

    const float* __restrict__ xg  = p.x;
    const float* __restrict__ gam = p.gamma[g] + bg * D;
    const float* __restrict__ bet = p.beta[g]  + bg * D;
    const float* __restrict__ Wg  = p.W[g]     + (size_t)bg * EMBD * D;
    const float* __restrict__ bsg = p.bias[g]  + bg * EMBD;

    // ---------------- Phase 1: gather + LayerNorm (8 threads per row) -------
    const int rowg = mtile * RPB + rloc;       // global row in [0,8000)
    const int b    = rowg / TT;
    const int t    = rowg - b * TT;

    // feature d = ((c*2)+r)*w + wi ; r=0/1 are adjacent in x -> float2 loads
    float s = 0.f, s2 = 0.f;
    for (int q = l8; q < 2 * w; q += 8) {      // q = (c,wi), c = q>=w (C==2)
        const int c  = (q >= w) ? 1 : 0;
        const int wi = q - c * w;
        const int f  = fstart + bg * w + wi;
        const float2 xv =
            *(const float2*)&xg[(((size_t)(b * CC + c) * FFREQ + f) * TT + t) * RR];
        const int d0 = (2 * c) * w + wi;       // r = 0
        const int d1 = d0 + w;                 // r = 1
        s_A[rloc * Kp + d0] = (_Float16)xv.x;
        s_A[rloc * Kp + d1] = (_Float16)xv.y;
        s  += xv.x + xv.y;
        s2 += xv.x * xv.x + xv.y * xv.y;
    }
    #pragma unroll
    for (int m = 1; m < 8; m <<= 1) {          // reduce within 8-lane row group
        s  += __shfl_xor(s,  m);
        s2 += __shfl_xor(s2, m);
    }
    const float mu = s / (float)D;
    const float rs = rsqrtf(s2 / (float)D - mu * mu + EPSV);

    // normalize in place (same thread owns the same slots -> no barrier)
    for (int q = l8; q < 2 * w; q += 8) {
        const int c  = (q >= w) ? 1 : 0;
        const int wi = q - c * w;
        const int d0 = (2 * c) * w + wi;
        const int d1 = d0 + w;
        const float v0 = (float)s_A[rloc * Kp + d0];
        const float v1 = (float)s_A[rloc * Kp + d1];
        s_A[rloc * Kp + d0] = (_Float16)((v0 - mu) * rs * gam[d0] + bet[d0]);
        s_A[rloc * Kp + d1] = (_Float16)((v1 - mu) * rs * gam[d1] + bet[d1]);
    }
    for (int k = D + l8; k < Kp; k += 8)       // zero-pad (band 60: D=4 -> 32)
        s_A[rloc * Kp + k] = (_Float16)0.f;

    // ---------------- Phase 2: WMMA GEMM over K ----------------
    v8f acc0 = {}, acc1 = {};
    const int nloc = lane & 15;
    const int hh   = lane >> 4;
    const int nglb = wave * 16 + nloc;         // output column e in [0,128)

    for (int ss = 0; ss < nsteps; ++ss) {
        __syncthreads();   // s_A ready / previous s_B consumed
        // Stage B[k][n] = W[n][k] for k in [32ss,32ss+32): branchless f32->f16.
        // Clamped b128 loads (always in-bounds), per-element cndmask zeroing,
        // packed b64 LDS stores. 4 independent loads per thread, unrolled.
        #pragma unroll
        for (int ii = 0; ii < 4; ++ii) {
            const int grp = tid + ii * 256;    // 0..1023 float4 groups
            const int n   = grp >> 3;          // 0..127
            const int kk  = (grp & 7) * 4;     // 0,4,..,28
            const int k   = ss * 32 + kk;
            const float4 wv =
                *(const float4*)&Wg[n * D + ((k + 3 < D) ? k : 0)];
            Pack4 pk;
            pk.h[0] = (_Float16)((k + 0 < D) ? wv.x : 0.f);
            pk.h[1] = (_Float16)((k + 1 < D) ? wv.y : 0.f);
            pk.h[2] = (_Float16)((k + 2 < D) ? wv.z : 0.f);
            pk.h[3] = (_Float16)((k + 3 < D) ? wv.w : 0.f);
            *(uint2*)&s_B[n * 32 + kk] = pk.u2;
        }
        // prefetch next K-chunk of W into cache (gfx1250 global_prefetch_b8)
        if (ss + 1 < nsteps)
            __builtin_prefetch(&Wg[(tid >> 3) * D + ss * 32 + 32 + (tid & 7) * 4], 0, 0);
        __syncthreads();

        // B fragment: 32x16 f16, lanes 0-15 K=0..7,16..23, lanes 16-31 K=8..15,24..31
        FragH bf;
        #pragma unroll
        for (int v = 0; v < 8; ++v)
            bf.u[v] = *(const unsigned*)&s_B[nglb * 32 + ((v & 4) ? 16 : 0) + hh * 8 + (v & 3) * 2];
        // A fragments (two 16-row M-tiles): 16x32 f16 layout
        FragH a0, a1;
        #pragma unroll
        for (int v = 0; v < 8; ++v) {
            const int ka = ss * 32 + ((v & 4) ? 16 : 0) + hh * 8 + (v & 3) * 2;
            a0.u[v] = *(const unsigned*)&s_A[nloc * Kp + ka];
            a1.u[v] = *(const unsigned*)&s_A[(16 + nloc) * Kp + ka];
        }
        acc0 = __builtin_amdgcn_wmma_f32_16x16x32_f16(
            false, a0.v, false, bf.v, (short)0, acc0, false, false);
        acc1 = __builtin_amdgcn_wmma_f32_16x16x32_f16(
            false, a1.v, false, bf.v, (short)0, acc1, false, false);
    }

    // ---------------- Phase 3: bias + store ----------------
    const float bv = bsg[nglb];
    #pragma unroll
    for (int r = 0; r < 8; ++r) {
        const int m = r + 8 * hh;              // C/D layout: VGPR r -> M row
        {   // tile 0
            const int rg = mtile * RPB + m;
            const int ob = rg / TT;
            const int ot = rg - ob * TT;
            p.out[(((size_t)ob * NBANDS + band) * TT + ot) * EMBD + nglb] =
                acc0[r] + bv;
        }
        {   // tile 1
            const int rg = mtile * RPB + 16 + m;
            const int ob = rg / TT;
            const int ot = rg - ob * TT;
            p.out[(((size_t)ob * NBANDS + band) * TT + ot) * EMBD + nglb] =
                acc1[r] + bv;
        }
    }
}

extern "C" void kernel_launch(void* const* d_in, const int* in_sizes, int n_in,
                              void* d_out, int out_size, void* d_ws, size_t ws_size,
                              hipStream_t stream) {
    (void)in_sizes; (void)n_in; (void)out_size; (void)d_ws; (void)ws_size;
    BSParams p;
    p.x = (const float*)d_in[0];
    for (int i = 0; i < 5; ++i) p.gamma[i] = (const float*)d_in[1 + i];
    for (int i = 0; i < 5; ++i) p.beta[i]  = (const float*)d_in[6 + i];
    for (int i = 0; i < 5; ++i) p.W[i]     = (const float*)d_in[11 + i];
    for (int i = 0; i < 5; ++i) p.bias[i]  = (const float*)d_in[16 + i];
    p.out = (float*)d_out;

    dim3 grid(MROWS / RPB, NBANDS);   // 250 x 61 = 15250 blocks
    hipLaunchKernelGGL(bandsplit_wmma_kernel, grid, dim3(256), 0, stream, p);
}
